// E83CircularTowerCell_76871324664644
// MI455X (gfx1250) — matrix operand order repaired and checked by reading, not compiled, not run on hardware
//
#include <hip/hip_runtime.h>
#include <hip/hip_bf16.h>
#include <math.h>

// ---------------- types for WMMA ----------------
typedef __bf16 bf16_t;
typedef bf16_t   v16bf __attribute__((ext_vector_type(16)));
typedef float    v8f   __attribute__((ext_vector_type(8)));
typedef unsigned v8u   __attribute__((ext_vector_type(8)));

// Pack two f32 into one dword of packed bf16 (lo -> bits[15:0], hi -> bits[31:16]).
// round-half-up via +0x8000 on the raw bits, then a single v_perm_b32 grabs the
// two high halves: sel bytes {7,6,3,2} of concat(src0=hi, src1=lo).
__device__ __forceinline__ unsigned pack2_bf16(float lo, float hi) {
  unsigned a = __builtin_bit_cast(unsigned, lo) + 0x8000u;
  unsigned b = __builtin_bit_cast(unsigned, hi) + 0x8000u;
  return __builtin_amdgcn_perm(b, a, 0x07060302u);
}

// =====================================================================
// Phase 1: P[row, col] = x[row,:] . W[col,:]   (row = t*B+b, col in [0,448))
//   cols 0..383 -> W_kv rows, cols 384..447 -> W_q rows
// One wave computes a 16(M) x 64(N) strip: 4 accumulators, A reused 4x.
// v_wmma_f32_16x16x32_bf16, f32 accumulate.
// =====================================================================
__global__ __launch_bounds__(256)
void proj_gemm(const float* __restrict__ x,
               const float* __restrict__ Wkv,
               const float* __restrict__ Wq,
               float* __restrict__ P)
{
  const int D = 1024, NC = 448;
  const int wave = (int)((blockIdx.x * blockDim.x + threadIdx.x) >> 5);
  const int lane = (int)(threadIdx.x & 31);
  const int hf   = lane >> 4;
  const int lm   = lane & 15;

  const int rowTile = wave / 7;      // 0..2047
  const int colGrp  = wave % 7;      // 0..6
  const int row0 = rowTile * 16;

  v8f acc[4];
#pragma unroll
  for (int c = 0; c < 4; ++c)
#pragma unroll
    for (int e = 0; e < 8; ++e) acc[c][e] = 0.0f;

  const float* wrow[4];
#pragma unroll
  for (int c = 0; c < 4; ++c) {
    int col = colGrp * 64 + c * 16 + lm;
    wrow[c] = (col < 384) ? (Wkv + (size_t)col * D)
                          : (Wq  + (size_t)(col - 384) * D);
  }
  const float* xrow = x + (size_t)(row0 + lm) * D;

  for (int kb = 0; kb < D; kb += 32) {
    // ---- A fragment: 16x32 bf16, ISA 16-bit A layout (pair packing) ----
    v8u au;
#pragma unroll
    for (int v = 0; v < 8; ++v) {
      int kk = ((v & 3) << 1) + ((v >> 2) << 4) + (hf << 3);
      float2 f = *(const float2*)(xrow + kb + kk);
      au[v] = pack2_bf16(f.x, f.y);
    }
    v16bf a = __builtin_bit_cast(v16bf, au);
#pragma unroll
    for (int c = 0; c < 4; ++c) {
      // ---- B fragment: 32x16 bf16, same K order as A (lanes 0-15 K=0..7,16..23; lanes 16-31 K=8..15,24..31) ----
      v8u bu;
#pragma unroll
      for (int v = 0; v < 8; ++v) {
        int kk = ((v & 3) << 1) + ((v >> 2) << 4) + (hf << 3);
        float2 f = *(const float2*)(wrow[c] + kb + kk);
        bu[v] = pack2_bf16(f.x, f.y);
      }
      v16bf bm = __builtin_bit_cast(v16bf, bu);
      acc[c] = __builtin_amdgcn_wmma_f32_16x16x32_bf16(
                   false, a, false, bm, (short)0, acc[c], false, false);
    }
  }

  // ---- store D: VGPR g holds M=g (lanes 0-15) / M=8+g (lanes 16-31), N=lm ----
#pragma unroll
  for (int c = 0; c < 4; ++c) {
    int col = colGrp * 64 + c * 16 + lm;
#pragma unroll
    for (int g = 0; g < 8; ++g) {
      int row = row0 + g + hf * 8;
      P[(size_t)row * NC + col] = acc[c][g];
    }
  }
}

// =====================================================================
// Phase 2: sequential scan. One block per batch b (16 blocks, 384 thr).
// M state (3 towers of 64x64 f32) lives in LDS with +1 row padding so
// both row (stride NP) and column (contiguous) accesses are conflict-free.
// The 3 matvecs are split 2-way over j across thread-halves.
// =====================================================================
#define KT 3
#define NN 64
#define NP 65
#define SCAN_THREADS 384

__global__ __launch_bounds__(SCAN_THREADS)
void tower_scan(const float* __restrict__ P,
                const float* __restrict__ M_init,
                const float* __restrict__ Bg,
                float* __restrict__ out,   // [T*B*NN] outputs, then M_final
                int T, int B)
{
  __shared__ float Msh[KT * NN * NP];   // 48.75 KB
  __shared__ float kn[KT * NN];
  __shared__ float vv[KT * NN];
  __shared__ float qv[NN];
  __shared__ float rowg[KT * NN];
  __shared__ float colg[KT * NN];
  __shared__ float dl[KT * NN];
  __shared__ float red[8];
  __shared__ float prv[SCAN_THREADS];
  __shared__ float pcv[SCAN_THREADS];
  __shared__ float prt[SCAN_THREADS];

  const int b   = (int)blockIdx.x;
  const int tid = (int)threadIdx.x;
  const int hlf = tid / 192;           // j-half for matvec split
  const int r   = tid - hlf * 192;     // (tower,row) id 0..191
  const int kt  = r / NN;
  const int ii  = r % NN;
  const int NC  = 448;

  // load M_init (K,B,n,n) into padded LDS
  for (int idx = tid; idx < KT * NN * NN; idx += SCAN_THREADS) {
    int k = idx >> 12, i = (idx >> 6) & 63, j = idx & 63;
    Msh[k * NN * NP + i * NP + j] =
        M_init[(((size_t)k * B + b) * NN + i) * NN + j];
  }
  const float bgate = (tid < KT * NN) ? Bg[kt * NN + ii] : 0.0f;
  __syncthreads();

  for (int t = 0; t < T; ++t) {
    const float* Pr = P + ((size_t)t * B + b) * NC;

    // ---- load k, v, q; wave-local |k|^2 reduce (wave32 shfl) ----
    float kraw = 0.0f;
    if (tid < KT * NN) {
      kraw    = Pr[kt * (2 * NN) + ii];        // k for tower kt
      vv[tid] = Pr[kt * (2 * NN) + NN + ii];   // v for tower kt
      float ss = kraw * kraw;
#pragma unroll
      for (int m = 16; m >= 1; m >>= 1) ss += __shfl_xor(ss, m, 32);
      if ((tid & 31) == 0) red[tid >> 5] = ss;
    } else if (tid < KT * NN + NN) {
      qv[tid - KT * NN] = Pr[6 * NN + (tid - KT * NN)];
    } else if (t + 4 < T && tid < KT * NN + NN + 14) {
      // prefetch the k/v/q row 4 steps ahead (448 floats = 14 x 128B)
      const float* Pn = P + ((size_t)(t + 4) * B + b) * NC;
      __builtin_prefetch(Pn + (tid - (KT * NN + NN)) * 32, 0, 1);
    }
    __syncthreads();

    if (tid < KT * NN) {
      float ss = red[2 * kt] + red[2 * kt + 1];
      kn[tid] = kraw / (sqrtf(ss) + 1e-6f);
    }
    __syncthreads();

    // ---- three matvecs per (tower,row), j split across two halves ----
    {
      int g = kt + 1; if (g == KT) g = 0;      // roll(M, -1)
      const float* Mg  = Msh + g  * NN * NP;
      const float* Mk  = Msh + kt * NN * NP;
      const float* knk = kn + kt * NN;
      const int j0 = hlf * (NN / 2);
      float rv = 0.0f, cv = 0.0f, rt = 0.0f;
#pragma unroll 8
      for (int j = j0; j < j0 + NN / 2; ++j) {
        float kj = knk[j];
        rv = fmaf(Mg[ii * NP + j], kj, rv);    // gater @ kn
        cv = fmaf(Mg[j * NP + ii], kj, cv);    // gaterᵀ @ kn
        rt = fmaf(Mk[ii * NP + j], kj, rt);    // M @ kn
      }
      prv[tid] = rv; pcv[tid] = cv; prt[tid] = rt;
    }
    __syncthreads();

    if (tid < KT * NN) {
      float rv = prv[tid] + prv[tid + 192];
      float cv = pcv[tid] + pcv[tid + 192];
      float rt = prt[tid] + prt[tid + 192];
      rowg[tid] = 1.0f / (1.0f + __expf(-(rv + bgate)));
      colg[tid] = 1.0f / (1.0f + __expf(-(cv + bgate)));
      dl[tid]   = vv[tid] - rt;
    }
    __syncthreads();

    // ---- gated rank-1 update of all 3*64*64 state elements ----
    for (int idx = tid; idx < KT * NN * NN; idx += SCAN_THREADS) {
      int k = idx >> 12, i = (idx >> 6) & 63, j = idx & 63;
      int s = k * NN * NP + i * NP + j;
      Msh[s] = rowg[k * NN + i] * Msh[s] * colg[k * NN + j]
             + dl[k * NN + i] * kn[k * NN + j];
    }
    __syncthreads();

    // ---- Sq = M_new[0] @ q ; out = SiLU(Sq) ----
    if (tid < NN) {
      float sq = 0.0f;
#pragma unroll 8
      for (int j = 0; j < NN; ++j) sq = fmaf(Msh[tid * NP + j], qv[j], sq);
      out[((size_t)t * B + b) * NN + tid] = sq / (1.0f + __expf(-sq));
    }
    __syncthreads();   // protect qv/red/kn/partials before next overwrite
  }

  // ---- write M_final (K,B,n,n) after outputs ----
  float* Mout = out + (size_t)T * B * NN;
  for (int idx = tid; idx < KT * NN * NN; idx += SCAN_THREADS) {
    int k = idx >> 12, i = (idx >> 6) & 63, j = idx & 63;
    Mout[(((size_t)k * B + b) * NN + i) * NN + j] = Msh[k * NN * NP + i * NP + j];
  }
}

// =====================================================================
extern "C" void kernel_launch(void* const* d_in, const int* in_sizes, int n_in,
                              void* d_out, int out_size, void* d_ws, size_t ws_size,
                              hipStream_t stream) {
  const float* x      = (const float*)d_in[0];  // (2048,16,1024)
  const float* M_init = (const float*)d_in[1];  // (3,16,64,64)
  const float* Wkv    = (const float*)d_in[2];  // (384,1024)
  const float* Wq     = (const float*)d_in[3];  // (64,1024)
  const float* Bgt    = (const float*)d_in[4];  // (3,64)
  float* out = (float*)d_out;
  float* P   = (float*)d_ws;                    // (32768, 448) f32 = 56 MB

  const int T = 2048, B = 16;

  // 2048 row-tiles * 7 col-groups = 14336 waves -> 1792 blocks of 8 waves
  proj_gemm<<<1792, 256, 0, stream>>>(x, Wkv, Wq, P);
  tower_scan<<<16, SCAN_THREADS, 0, stream>>>(P, M_init, Bgt, out, T, B);
}
